// LongNetViT_35132832481419
// MI455X (gfx1250) — matrix operand order repaired and checked by reading, not compiled, not run on hardware
//
#include <hip/hip_runtime.h>
#include <hip/hip_bf16.h>
#include <math.h>

typedef __attribute__((ext_vector_type(16))) __bf16 v16bf;
typedef __attribute__((ext_vector_type(8)))  float  v8f;

constexpr int   D    = 256;
constexpr int   H    = 8;
constexpr int   HD   = 32;
constexpr int   S    = 8192;
constexpr int   FFN  = 1024;
constexpr int   DIN  = 1536;
constexpr float SCL  = 0.17677669529663687f;   // 32^-0.5

// Fill a 16-element half of an A-style fragment from two float4 runs.
static __device__ __forceinline__ void pack8(v16bf& f, int base,
                                             const float4 a, const float4 b) {
  f[base + 0] = (__bf16)a.x; f[base + 1] = (__bf16)a.y;
  f[base + 2] = (__bf16)a.z; f[base + 3] = (__bf16)a.w;
  f[base + 4] = (__bf16)b.x; f[base + 5] = (__bf16)b.y;
  f[base + 6] = (__bf16)b.z; f[base + 7] = (__bf16)b.w;
}

// Load a 16x32 A-fragment row slice (two 8-wide runs) from a row pointer.
// A-layout: lane<16 holds K {0..15}, lane>=16 holds {16..31}.
static __device__ __forceinline__ v16bf load_afrag(const float* rp, int hi) {
  const float4* p0 = (const float4*)(rp + hi * 16);
  const float4* p1 = (const float4*)(rp + hi * 16 + 8);
  v16bf f;
  pack8(f, 0, p0[0], p0[1]);
  pack8(f, 8, p1[0], p1[1]);
  return f;
}

// ---------------------------------------------------------------------------
// WMMA GEMM: C[M,N] = act(A[M,K] @ B[K,N] + bias) + res
// One 64x16 output tile per wave: 4 M-subtiles share each B fragment,
// K-loop of v_wmma_f32_16x16x32_bf16.  act: 0 = none, 1 = exact GELU.
// ---------------------------------------------------------------------------
__global__ void __launch_bounds__(128)
gemm_wmma_kernel(const float* __restrict__ A, const float* __restrict__ B,
                 const float* __restrict__ bias, const float* __restrict__ res,
                 float* __restrict__ C, int M, int N, int K, int act)
{
  const int lane = threadIdx.x & 31;
  const int hi   = lane >> 4;
  const int l16  = lane & 15;
  const int nt_n = N >> 4;
  const int wave = blockIdx.x * (blockDim.x >> 5) + (threadIdx.x >> 5);
  const int mt = wave / nt_n;            // 64-row block index
  const int nt = wave - mt * nt_n;
  if (mt * 64 >= M) return;

  // Clamped A row pointers (rows >= M only feed never-stored D rows).
  const float* Ap[4];
  #pragma unroll
  for (int sub = 0; sub < 4; ++sub) {
    int r = mt * 64 + sub * 16 + l16;
    Ap[sub] = A + (size_t)(r < M ? r : M - 1) * K;
  }
  const int col = nt * 16 + l16;
  const float* Bp = B + col;

  v8f acc[4] = {};
  for (int kb = 0; kb < K; kb += 32) {
    __builtin_prefetch(Ap[0] + kb + 128, 0, 0);
    // B fragment (32x16): col = lane&15, K = hi*16 + h  (strided b32 loads)
    v16bf bfm;
    const float* Bk = Bp + (size_t)(kb + hi * 16) * N;
    #pragma unroll
    for (int h = 0; h < 16; ++h) bfm[h] = (__bf16)Bk[(size_t)h * N];
    #pragma unroll
    for (int sub = 0; sub < 4; ++sub) {
      v16bf af = load_afrag(Ap[sub] + kb, hi);
      acc[sub] = __builtin_amdgcn_wmma_f32_16x16x32_bf16(false, af, false, bfm,
                                                         (short)0, acc[sub],
                                                         false, false);
    }
  }
  #pragma unroll
  for (int sub = 0; sub < 4; ++sub) {
    #pragma unroll
    for (int v = 0; v < 8; ++v) {
      int row = mt * 64 + sub * 16 + hi * 8 + v;   // D: row = vgpr + 8*half
      if (row < M) {
        float x = acc[sub][v];
        if (bias) x += bias[col];
        if (act == 1) x = 0.5f * x * (1.0f + erff(x * 0.70710678118654752f));
        if (res)  x += res[(size_t)row * N + col];
        C[(size_t)row * N + col] = x;
      }
    }
  }
}

// ---------------------------------------------------------------------------
// Positional embedding + CLS token. tab[0] == 0, so row 0 is just cls_tok.
// ---------------------------------------------------------------------------
__global__ void pos_cls_kernel(float* __restrict__ hbuf,
                               const int* __restrict__ coords,
                               const float* __restrict__ cls)
{
  int s = blockIdx.x, d = threadIdx.x;
  if (s == 0) { hbuf[d] = cls[d]; return; }
  int t  = s - 1;
  int g0 = coords[t * 2 + 0] / 256;
  int g1 = coords[t * 2 + 1] / 256;
  int quad = d >> 6, j = d & 63;
  float omega = __expf(-(float)j * (9.210340371976184f / 64.0f)); // 10000^(-j/64)
  float p = (float)((quad < 2) ? g1 : g0);
  float o = p * omega;
  float val = ((quad & 1) == 0) ? sinf(o) : cosf(o);
  hbuf[(size_t)s * D + d] += val;
}

// ---------------------------------------------------------------------------
// LayerNorm: one wave per 256-wide row, float4 loads, wave32 shfl reduction.
// ---------------------------------------------------------------------------
__global__ void __launch_bounds__(128)
ln_kernel(const float* __restrict__ in, const float* __restrict__ g,
          const float* __restrict__ b, float* __restrict__ out, int rows)
{
  int row = blockIdx.x * 4 + (threadIdx.x >> 5);
  if (row >= rows) return;
  int lane = threadIdx.x & 31;
  const float4* x4 = (const float4*)(in + (size_t)row * D + lane * 8);
  float4 v0 = x4[0], v1 = x4[1];
  float s  = v0.x + v0.y + v0.z + v0.w + v1.x + v1.y + v1.z + v1.w;
  float sq = v0.x*v0.x + v0.y*v0.y + v0.z*v0.z + v0.w*v0.w
           + v1.x*v1.x + v1.y*v1.y + v1.z*v1.z + v1.w*v1.w;
  #pragma unroll
  for (int d = 1; d < 32; d <<= 1) {
    s  += __shfl_xor(s, d, 32);
    sq += __shfl_xor(sq, d, 32);
  }
  float mu = s * (1.0f / D);
  float rs = rsqrtf(sq * (1.0f / D) - mu * mu + 1e-5f);
  const float4* g4 = (const float4*)(g + lane * 8);
  const float4* b4 = (const float4*)(b + lane * 8);
  float4 g0 = g4[0], g1 = g4[1], b0 = b4[0], b1 = b4[1];
  float4 y0, y1;
  y0.x = (v0.x-mu)*rs*g0.x+b0.x; y0.y = (v0.y-mu)*rs*g0.y+b0.y;
  y0.z = (v0.z-mu)*rs*g0.z+b0.z; y0.w = (v0.w-mu)*rs*g0.w+b0.w;
  y1.x = (v1.x-mu)*rs*g1.x+b1.x; y1.y = (v1.y-mu)*rs*g1.y+b1.y;
  y1.z = (v1.z-mu)*rs*g1.z+b1.z; y1.w = (v1.w-mu)*rs*g1.w+b1.w;
  float4* o4 = (float4*)(out + (size_t)row * D + lane * 8);
  o4[0] = y0; o4[1] = y1;
}

// ---------------------------------------------------------------------------
// Init per-config attention buffers: O = 0, lse = -1e30.
// ---------------------------------------------------------------------------
__global__ void attn_init_kernel(float* __restrict__ O, float* __restrict__ L)
{
  size_t i = (size_t)blockIdx.x * blockDim.x + threadIdx.x;
  if (i < (size_t)5 * S * D) O[i] = 0.0f;
  if (i < (size_t)5 * S * H) L[i] = -1e30f;
}

// ---------------------------------------------------------------------------
// Dilated attention: every (config,segment,head) is a 1024-token attention
// over strided tokens tok(j) = n*w + (head mod r) + j*r (t = w/r = 1024 for
// all configs).  Grid: 16 cfg-segments * 8 heads * 16 qtiles = 2048 blocks
// of 4 waves; each wave flash-attends 16 queries.  All loads are clamped
// (no divergent guards); masking happens arithmetically on the scores.
// ---------------------------------------------------------------------------
__global__ void __launch_bounds__(128)
attn_kernel(const float* __restrict__ q, const float* __restrict__ k,
            const float* __restrict__ v, float* __restrict__ O,
            float* __restrict__ L)
{
  __shared__ float sP[4][16][32];       // per-wave P-tile staging (D->A relayout)
  const int bid   = blockIdx.x;
  const int qtile = bid & 15;
  const int head  = (bid >> 4) & 7;
  const int cs    = bid >> 7;           // 0..15 -> (config, segment)
  int c, n;
  if      (cs < 8)  { c = 0; n = cs;      }
  else if (cs < 12) { c = 1; n = cs - 8;  }
  else if (cs < 14) { c = 2; n = cs - 12; }
  else if (cs < 15) { c = 3; n = 0;       }
  else              { c = 4; n = 0;       }
  const int r    = 1 << c;
  const int base = n * (1024 << c) + (head & (r - 1));

  const int wid  = threadIdx.x >> 5;
  const int lane = threadIdx.x & 31;
  const int hi   = lane >> 4;
  const int l16  = lane & 15;

  // Q fragment: 16 queries x 32 head-dims (one WMMA K-step); clamped row.
  const int qslot = qtile * 64 + wid * 16 + l16;
  const int sq    = base + (qslot << c);
  const v16bf qf  = load_afrag(q + (size_t)(sq < S ? sq : S - 1) * D + head * HD, hi);

  float mrun[8], lrun[8];
  #pragma unroll
  for (int i = 0; i < 8; ++i) { mrun[i] = -1e30f; lrun[i] = 0.0f; }
  v8f o0 = {}, o1 = {};

  for (int j0 = 0; j0 < 1024; j0 += 32) {
    v8f s0 = {}, s1 = {};
    #pragma unroll
    for (int p = 0; p < 2; ++p) {           // two 16-key score tiles
      int sk = base + ((j0 + p * 16 + l16) << c);
      // B fragment: col = key (lane&15), K = head-dim hi*16 + h; 16 contiguous.
      const float4* kp = (const float4*)(k + (size_t)(sk < S ? sk : S - 1) * D
                                           + head * HD + hi * 16);
      float4 k0 = kp[0], k1 = kp[1], k2 = kp[2], k3 = kp[3];
      v16bf kf;
      pack8(kf, 0, k0, k1);
      pack8(kf, 8, k2, k3);
      if (p == 0)
        s0 = __builtin_amdgcn_wmma_f32_16x16x32_bf16(false, qf, false, kf,
                                                     (short)0, s0, false, false);
      else
        s1 = __builtin_amdgcn_wmma_f32_16x16x32_bf16(false, qf, false, kf,
                                                     (short)0, s1, false, false);
    }
    const bool ok0 = (base + ((j0 + l16)      << c)) < S;
    const bool ok1 = (base + ((j0 + 16 + l16) << c)) < S;
    float p0[8], p1[8];
    #pragma unroll
    for (int vv = 0; vv < 8; ++vv) {
      float a0 = ok0 ? s0[vv] * SCL : -1e9f;   // arithmetic mask, no branch
      float a1 = ok1 ? s1[vv] * SCL : -1e9f;
      float mv = fmaxf(a0, a1);                // row max over 32 keys
      mv = fmaxf(mv, __shfl_xor(mv, 1, 32));
      mv = fmaxf(mv, __shfl_xor(mv, 2, 32));
      mv = fmaxf(mv, __shfl_xor(mv, 4, 32));
      mv = fmaxf(mv, __shfl_xor(mv, 8, 32));
      float mn = fmaxf(mrun[vv], mv);
      float e  = __expf(mrun[vv] - mn);
      p0[vv] = __expf(a0 - mn);
      p1[vv] = __expf(a1 - mn);
      float ps = p0[vv] + p1[vv];              // row sum
      ps += __shfl_xor(ps, 1, 32);
      ps += __shfl_xor(ps, 2, 32);
      ps += __shfl_xor(ps, 4, 32);
      ps += __shfl_xor(ps, 8, 32);
      lrun[vv] = lrun[vv] * e + ps;
      mrun[vv] = mn;
      o0[vv] *= e; o1[vv] *= e;
    }
    // stage P (D-fragment layout) -> LDS -> reload as A-fragment (16x32)
    #pragma unroll
    for (int vv = 0; vv < 8; ++vv) {
      sP[wid][hi * 8 + vv][l16]      = p0[vv];
      sP[wid][hi * 8 + vv][16 + l16] = p1[vv];
    }
    v16bf pf;
    #pragma unroll
    for (int h = 0; h < 16; ++h) {
      int kk = hi * 16 + h;
      pf[h] = (__bf16)sP[wid][l16][kk];
    }
    // V B-fragments for output dims [0:16) and [16:32); clamped rows.
    v16bf vf0, vf1;
    #pragma unroll
    for (int h = 0; h < 16; ++h) {
      int sk = base + ((j0 + hi * 16 + h) << c);
      const float* vp = v + (size_t)(sk < S ? sk : S - 1) * D + head * HD + l16;
      vf0[h] = (__bf16)vp[0];
      vf1[h] = (__bf16)vp[16];
    }
    o0 = __builtin_amdgcn_wmma_f32_16x16x32_bf16(false, pf, false, vf0,
                                                 (short)0, o0, false, false);
    o1 = __builtin_amdgcn_wmma_f32_16x16x32_bf16(false, pf, false, vf1,
                                                 (short)0, o1, false, false);
  }
  // epilogue: normalize, scatter to per-config O, write lse
  #pragma unroll
  for (int vv = 0; vv < 8; ++vv) {
    int qs  = qtile * 64 + wid * 16 + hi * 8 + vv;
    int sq2 = base + (qs << c);
    if (sq2 < S) {
      float inv = 1.0f / lrun[vv];
      size_t ob = ((size_t)c * S + sq2) * D + head * HD;
      O[ob + l16]      = o0[vv] * inv;
      O[ob + 16 + l16] = o1[vv] * inv;
      if (l16 == 0)
        L[((size_t)c * S + sq2) * H + head] = mrun[vv] + logf(lrun[vv]);
    }
  }
}

// ---------------------------------------------------------------------------
// LSE-weighted combine of the 5 dilated-attention branches.
// ---------------------------------------------------------------------------
__global__ void combine_kernel(const float* __restrict__ O,
                               const float* __restrict__ L,
                               float* __restrict__ out)
{
  int s = blockIdx.x, d = threadIdx.x, hh = d >> 5;
  float Ls[5], m = -1e38f;
  #pragma unroll
  for (int c = 0; c < 5; ++c) {
    Ls[c] = L[((size_t)c * S + s) * H + hh];
    m = fmaxf(m, Ls[c]);
  }
  float sw = 0.f, acc = 0.f;
  #pragma unroll
  for (int c = 0; c < 5; ++c) {
    float w = __expf(Ls[c] - m);
    sw  += w;
    acc += w * O[((size_t)c * S + s) * D + hh * HD + (d & 31)];
  }
  out[(size_t)s * D + d] = acc / sw;
}

// ---------------------------------------------------------------------------
// Final double-LayerNorm of the CLS row (row 0), single wave.
// ---------------------------------------------------------------------------
__global__ void final_kernel(const float* __restrict__ hbuf,
                             const float* __restrict__ eg, const float* __restrict__ eb,
                             const float* __restrict__ ng, const float* __restrict__ nb,
                             float* __restrict__ out)
{
  int lane = threadIdx.x & 31;
  float x[8];
  #pragma unroll
  for (int i = 0; i < 8; ++i) x[i] = hbuf[lane * 8 + i];
  #pragma unroll
  for (int pass = 0; pass < 2; ++pass) {
    float s = 0.f, sq = 0.f;
    #pragma unroll
    for (int i = 0; i < 8; ++i) { s += x[i]; sq += x[i] * x[i]; }
    #pragma unroll
    for (int d = 1; d < 32; d <<= 1) {
      s  += __shfl_xor(s, d, 32);
      sq += __shfl_xor(sq, d, 32);
    }
    float mu = s * (1.0f / D);
    float rs = rsqrtf(sq * (1.0f / D) - mu * mu + 1e-5f);
    const float* g = pass ? ng : eg;
    const float* b = pass ? nb : eb;
    #pragma unroll
    for (int i = 0; i < 8; ++i) {
      int idx = lane * 8 + i;
      x[i] = (x[i] - mu) * rs * g[idx] + b[idx];
    }
  }
  #pragma unroll
  for (int i = 0; i < 8; ++i) out[lane * 8 + i] = x[i];
}

// ---------------------------------------------------------------------------
extern "C" void kernel_launch(void* const* d_in, const int* in_sizes, int n_in,
                              void* d_out, int out_size, void* d_ws, size_t ws_size,
                              hipStream_t stream)
{
  const float* x      = (const float*)d_in[0];
  const int*   coords = (const int*)  d_in[1];
  const float* proj_w = (const float*)d_in[2];
  const float* proj_b = (const float*)d_in[3];
  const float* cls    = (const float*)d_in[4];
  const float* Wq     = (const float*)d_in[5];
  const float* Wk     = (const float*)d_in[6];
  const float* Wv     = (const float*)d_in[7];
  const float* Wo     = (const float*)d_in[8];
  const float* bq     = (const float*)d_in[9];
  const float* bk     = (const float*)d_in[10];
  const float* bv     = (const float*)d_in[11];
  const float* bo     = (const float*)d_in[12];
  const float* ln1g   = (const float*)d_in[13];
  const float* ln1b   = (const float*)d_in[14];
  const float* ln2g   = (const float*)d_in[15];
  const float* ln2b   = (const float*)d_in[16];
  const float* W1     = (const float*)d_in[17];
  const float* b1     = (const float*)d_in[18];
  const float* W2     = (const float*)d_in[19];
  const float* b2     = (const float*)d_in[20];
  const float* encg   = (const float*)d_in[21];
  const float* encb   = (const float*)d_in[22];
  const float* normg  = (const float*)d_in[23];
  const float* normb  = (const float*)d_in[24];

  float* ws = (float*)d_ws;
  float* hB = ws; ws += (size_t)S * D;        // residual stream
  float* aB = ws; ws += (size_t)S * D;        // LN output
  float* qB = ws; ws += (size_t)S * D;
  float* kB = ws; ws += (size_t)S * D;
  float* vB = ws; ws += (size_t)S * D;
  float* aO = ws; ws += (size_t)5 * S * D;    // per-config attn outputs
  float* aL = ws; ws += (size_t)5 * S * H;    // per-config lse
  float* aC = ws; ws += (size_t)S * D;        // combined attention
  float* fB = ws; ws += (size_t)S * FFN;      // FFN hidden

  auto gemm = [&](const float* A, const float* B, const float* bias,
                  const float* res, float* C, int M, int N, int K, int act) {
    int waves = ((M + 63) / 64) * (N / 16);
    gemm_wmma_kernel<<<(waves + 3) / 4, 128, 0, stream>>>(A, B, bias, res, C,
                                                          M, N, K, act);
  };

  // Projection into rows 1..8191; row 0 (cls) + pos-embed elementwise.
  gemm(x, proj_w, proj_b, nullptr, hB + D, S - 1, D, DIN, 0);
  pos_cls_kernel<<<S, D, 0, stream>>>(hB, coords, cls);

  for (int l = 0; l < 2; ++l) {
    ln_kernel<<<S / 4, 128, 0, stream>>>(hB, ln1g + l * D, ln1b + l * D, aB, S);
    gemm(aB, Wq + l * D * D, bq + l * D, nullptr, qB, S, D, D, 0);
    gemm(aB, Wk + l * D * D, bk + l * D, nullptr, kB, S, D, D, 0);
    gemm(aB, Wv + l * D * D, bv + l * D, nullptr, vB, S, D, D, 0);
    attn_init_kernel<<<(5 * S * D + 255) / 256, 256, 0, stream>>>(aO, aL);
    attn_kernel<<<2048, 128, 0, stream>>>(qB, kB, vB, aO, aL);
    combine_kernel<<<S, D, 0, stream>>>(aO, aL, aC);
    gemm(aC, Wo + l * D * D, bo + l * D, hB, hB, S, D, D, 0);
    ln_kernel<<<S / 4, 128, 0, stream>>>(hB, ln2g + l * D, ln2b + l * D, aB, S);
    gemm(aB, W1 + l * D * FFN, b1 + l * FFN, nullptr, fB, S, FFN, D, 1);
    gemm(fB, W2 + l * FFN * D, b2 + l * D, hB, hB, S, D, FFN, 0);
  }
  final_kernel<<<1, 32, 0, stream>>>(hB, encg, encb, normg, normb, (float*)d_out);
}
